// Encoder_26560077758766
// MI455X (gfx1250) — matrix operand order repaired and checked by reading, not compiled, not run on hardware
//
#include <hip/hip_runtime.h>
#include <hip/hip_bf16.h>

// GRU (1 layer, return last hidden) for MI455X / gfx1250, wave32 WMMA.
// BATCH=64, SEQ=1024, IN_DIM=512, HID=1024.
//
// Phase 1: gi[s][b][3H] = x @ W_ih^T + b_ih      (bf16 WMMA, f32 acc; parallel)
// Phase 2: ONE persistent kernel, 32 blocks (1 per WGP), 1024 steps separated
//          by a device-wide sense-reversing barrier.
//          - W_hh slice resident in LDS (192 KB) for all 1024 steps
//          - f32 h in registers (static (m,n) ownership), bf16 h ping-pongs
//          - h staging uses CDNA5 GLOBAL_LOAD_ASYNC_TO_LDS_B128 DMA,
//            double-buffered (2 x 32 KB) so the DMA of chunk c+1 overlaps the
//            24 WMMAs of chunk c. LDS total: 256 KB of the WGP's 320 KB.

#define BATCH  64
#define SEQ    1024
#define IN_DIM 512
#define HID    1024
#define G3     (3 * HID)   // 3072

typedef __bf16 bf16_t;
typedef __attribute__((ext_vector_type(16))) __bf16 v16bf;
typedef __attribute__((ext_vector_type(8)))  __bf16 v8bf;
typedef __attribute__((ext_vector_type(8)))  float  v8f;
typedef int v4i_vs __attribute__((vector_size(16)));   // matches builtin param

// ---- CDNA5 async global->LDS copy (ASYNCcnt path) ---------------------------
__device__ __forceinline__ void async_copy_b128(const bf16_t* gsrc, bf16_t* ldst) {
#if __has_builtin(__builtin_amdgcn_global_load_async_to_lds_b128)
    __builtin_amdgcn_global_load_async_to_lds_b128(
        (__attribute__((address_space(1))) v4i_vs*)gsrc,
        (__attribute__((address_space(3))) v4i_vs*)ldst, 0, 0);
#else
    unsigned lds_addr = (unsigned)(unsigned long long)
        (__attribute__((address_space(3))) bf16_t*)ldst;
    asm volatile("global_load_async_to_lds_b128 %0, %1, off"
                 :: "v"(lds_addr), "v"((unsigned long long)gsrc)
                 : "memory");
#endif
}

__device__ __forceinline__ void wait_asynccnt0() {
#if __has_builtin(__builtin_amdgcn_s_wait_asynccnt)
    __builtin_amdgcn_s_wait_asynccnt(0);
#else
    asm volatile("s_wait_asynccnt 0x0" ::: "memory");
#endif
}

// ---- fragment loaders -------------------------------------------------------
// A (16x32 bf16, MxK): lanes 0-15 hold row M=lane, K = kk+{0..7, 16..23};
//                      lanes 16-31 hold row M=lane-16, K = kk+{8..15, 24..31}.
__device__ __forceinline__ v16bf load_a_frag(const bf16_t* __restrict__ base,
                                             int lda, int mrow, int kk, int lane) {
    const int r  = lane & 15;
    const int hl = lane >> 4;
    const bf16_t* p = base + (size_t)(mrow + r) * lda + kk + 8 * hl;
    v8bf lo = *(const v8bf*)(p);
    v8bf hi = *(const v8bf*)(p + 16);
    v16bf a;
#pragma unroll
    for (int i = 0; i < 8; ++i) { a[i] = lo[i]; a[8 + i] = hi[i]; }
    return a;
}

// B (32x16 bf16, KxN), stored as Wt[n][k] row-major (ldb = K stride):
// lanes 0-15: column n = ncol+lane, K = kk+{0..7, 16..23};
// lanes 16-31: column n = ncol+lane-16, K = kk+{8..15, 24..31}.
__device__ __forceinline__ v16bf load_b_frag(const bf16_t* __restrict__ base,
                                             int ldb, int ncol, int kk, int lane) {
    const int c  = lane & 15;
    const int hl = lane >> 4;
    const bf16_t* p = base + (size_t)(ncol + c) * ldb + kk + 8 * hl;
    v8bf lo = *(const v8bf*)(p);
    v8bf hi = *(const v8bf*)(p + 16);
    v16bf b;
#pragma unroll
    for (int i = 0; i < 8; ++i) { b[i] = lo[i]; b[8 + i] = hi[i]; }
    return b;
}

// ---- device-wide barrier (persistent grid, 32 co-resident blocks) ----------
__device__ __forceinline__ void grid_barrier(unsigned* bar, unsigned nblocks) {
    __threadfence();          // release my stores to device scope
    __syncthreads();
    if (threadIdx.x == 0) {
        unsigned g = __hip_atomic_load(&bar[1], __ATOMIC_RELAXED,
                                       __HIP_MEMORY_SCOPE_AGENT);
        unsigned arrived = __hip_atomic_fetch_add(&bar[0], 1u, __ATOMIC_ACQ_REL,
                                                  __HIP_MEMORY_SCOPE_AGENT);
        if (arrived == nblocks - 1u) {
            __hip_atomic_store(&bar[0], 0u, __ATOMIC_RELAXED,
                               __HIP_MEMORY_SCOPE_AGENT);
            __hip_atomic_store(&bar[1], g + 1u, __ATOMIC_RELEASE,
                               __HIP_MEMORY_SCOPE_AGENT);
        } else {
            while (__hip_atomic_load(&bar[1], __ATOMIC_ACQUIRE,
                                     __HIP_MEMORY_SCOPE_AGENT) == g) {
                __builtin_amdgcn_s_sleep(2);
            }
        }
    }
    __syncthreads();
    __threadfence();          // acquire: discard stale cached lines
}

// ---- helpers ----------------------------------------------------------------
__global__ void f32_to_bf16_kernel(const float* __restrict__ src,
                                   bf16_t* __restrict__ dst, size_t n) {
    size_t i = (size_t)blockIdx.x * blockDim.x + threadIdx.x;
    if (i < n) dst[i] = (bf16_t)src[i];
}

__global__ void init_kernel(bf16_t* __restrict__ hb, unsigned* __restrict__ bar,
                            int n) {
    int i = blockIdx.x * blockDim.x + threadIdx.x;
    if (i < n) hb[i] = (bf16_t)0.0f;
    if (i < 2) bar[i] = 0u;
}

// ---- Phase 1: gi = x @ W_ih^T + b_ih ---------------------------------------
__global__ __launch_bounds__(256)
void gi_gemm_kernel(const bf16_t* __restrict__ xb,    // [B*S, I] (row = b*SEQ + s)
                    const bf16_t* __restrict__ wih,   // [3H, I]  (B operand)
                    const float*  __restrict__ b_ih,  // [3H]
                    float*        __restrict__ gi) {  // [S, B, 3H]
    const int lane = threadIdx.x & 31;
    const int wv   = threadIdx.x >> 5;
    const int mrow = blockIdx.x * 32 + (wv >> 2) * 16;       // over B*S
    const int ncol = blockIdx.y * 64 + (wv & 3) * 16;        // over 3H

    v8f acc = {};
#pragma unroll 4
    for (int kk = 0; kk < IN_DIM; kk += 32) {
        v16bf a = load_a_frag(xb,  IN_DIM, mrow, kk, lane);
        v16bf b = load_b_frag(wih, IN_DIM, ncol, kk, lane);
        acc = __builtin_amdgcn_wmma_f32_16x16x32_bf16(false, a, false, b,
                                                      (short)0, acc, false, false);
    }

    const int n    = ncol + (lane & 15);
    const int hl   = lane >> 4;
    const float bi = b_ih[n];
#pragma unroll
    for (int v = 0; v < 8; ++v) {
        int m  = mrow + v + 8 * hl;          // row = b*SEQ + s
        int s  = m & (SEQ - 1);
        int b_ = m >> 10;                    // SEQ == 1024
        gi[((size_t)s * BATCH + b_) * G3 + n] = acc[v] + bi;
    }
}

// ---- Phase 2: persistent GRU scan ------------------------------------------
// Grid = 32 blocks; block owns 32 columns of H for all 1024 steps.
// Wave w -> rows (w>>1)*16..+15, local col tile (w&1)*16; 3 gate accumulators.
__global__ __launch_bounds__(256)
void gru_scan_kernel(const float*  __restrict__ gi,     // [S, B, 3H]
                     const bf16_t* __restrict__ whh,    // [3H, HID] (B operand)
                     const float*  __restrict__ b_hh,   // [3H]
                     bf16_t* hAb, bf16_t* hBb,          // [BATCH, HID] bf16 ping-pong
                     float*  __restrict__ out,          // [BATCH, HID] final h
                     unsigned* bar) {
    __shared__ bf16_t wsh[96 * HID];        // 192 KB: rows = gate*32 + localcol
    __shared__ bf16_t hsh[2][BATCH * 256];  // 2 x 32 KB: double-buffered h chunks

    const int lane = threadIdx.x & 31;
    const int wv   = threadIdx.x >> 5;
    const int mrow = (wv >> 1) * 16;               // batch row tile
    const int jloc = (wv & 1) * 16;                // local col tile (0 or 16)
    const int j0   = blockIdx.x * 32;              // global H column base
    const int hl   = lane >> 4;
    const int n    = j0 + jloc + (lane & 15);      // global H column

    // ---- preload this block's W_hh slice into LDS (once, reused 1024x) ----
    for (int i = threadIdx.x; i < (96 * HID) / 8; i += 256) {
        int row = i >> 7;                 // 0..95
        int k8  = (i & 127) * 8;
        int g   = row >> 5;
        int c   = row & 31;
        *(v8bf*)&wsh[(size_t)row * HID + k8] =
            *(const v8bf*)&whh[((size_t)g * HID + j0 + c) * HID + k8];
    }
    __syncthreads();

    const float bhr = b_hh[n];
    const float bhz = b_hh[HID + n];
    const float bhn = b_hh[2 * HID + n];

    // f32 hidden state lives in registers: lane owns (m = mrow+v+8*hl, n).
    float hreg[8];
#pragma unroll
    for (int v = 0; v < 8; ++v) hreg[v] = 0.0f;

    for (int t = 0; t < SEQ; ++t) {
        const bf16_t* hpb = (t & 1) ? hBb : hAb;
        bf16_t*       hnb = (t & 1) ? hAb : hBb;
        const float*  gi_t = gi + (size_t)t * BATCH * G3;

        v8f acc[3] = {v8f{}, v8f{}, v8f{}};

        // issue async DMA stage of chunk 0 into buffer 0
#pragma unroll 2
        for (int i = threadIdx.x; i < (BATCH * 256) / 8; i += 256) {
            int row = i >> 5;
            int c8  = (i & 31) * 8;
            async_copy_b128(&hpb[(size_t)row * HID + c8], &hsh[0][row * 256 + c8]);
        }

        for (int chunk = 0; chunk < 4; ++chunk) {
            const int kbase = chunk * 256;
            wait_asynccnt0();        // my DMAs for chunk `chunk` are in LDS
            __syncthreads();         // everyone's DMAs are in LDS

            if (chunk < 3) {
                // overlap: DMA chunk+1 into the other buffer during compute.
                // That buffer was last read at chunk-1; all waves passed the
                // barrier above after finishing it, so no WAR hazard.
                const int kn = kbase + 256;
#pragma unroll 2
                for (int i = threadIdx.x; i < (BATCH * 256) / 8; i += 256) {
                    int row = i >> 5;
                    int c8  = (i & 31) * 8;
                    async_copy_b128(&hpb[(size_t)row * HID + kn + c8],
                                    &hsh[(chunk + 1) & 1][row * 256 + c8]);
                }
            }

            const bf16_t* hbuf = hsh[chunk & 1];
#pragma unroll 2
            for (int kk = 0; kk < 256; kk += 32) {
                v16bf a = load_a_frag(hbuf, 256, mrow, kk, lane);
#pragma unroll
                for (int g = 0; g < 3; ++g) {
                    v16bf b = load_b_frag(wsh, HID, g * 32 + jloc,
                                          kbase + kk, lane);
                    acc[g] = __builtin_amdgcn_wmma_f32_16x16x32_bf16(
                        false, a, false, b, (short)0, acc[g], false, false);
                }
            }
        }

        // fused gates; update register-resident f32 h; emit bf16 h for t+1
#pragma unroll
        for (int v = 0; v < 8; ++v) {
            int m = mrow + v + 8 * hl;
            const float* gr = gi_t + (size_t)m * G3;
            float ir  = gr[n];
            float iz  = gr[HID + n];
            float in_ = gr[2 * HID + n];
            float r  = 1.0f / (1.0f + __expf(-(ir + acc[0][v] + bhr)));
            float z  = 1.0f / (1.0f + __expf(-(iz + acc[1][v] + bhz)));
            float ng = tanhf(in_ + r * (acc[2][v] + bhn));
            float hv = (1.0f - z) * ng + z * hreg[v];
            hreg[v] = hv;
            hnb[(size_t)m * HID + n] = (bf16_t)hv;
            if (t == SEQ - 1) out[(size_t)m * HID + n] = hv;
        }

        grid_barrier(bar, gridDim.x);
    }
}

// ---- launcher ---------------------------------------------------------------
extern "C" void kernel_launch(void* const* d_in, const int* in_sizes, int n_in,
                              void* d_out, int out_size, void* d_ws, size_t ws_size,
                              hipStream_t stream) {
    (void)in_sizes; (void)n_in; (void)out_size; (void)ws_size;
    const float* x    = (const float*)d_in[0];
    const float* W_ih = (const float*)d_in[1];
    const float* W_hh = (const float*)d_in[2];
    const float* b_ih = (const float*)d_in[3];
    const float* b_hh = (const float*)d_in[4];
    float* out = (float*)d_out;

    char* ws = (char*)d_ws;
    size_t off = 0;
    auto alloc = [&](size_t bytes) -> void* {
        void* p = ws + off;
        off = (off + bytes + 255) & ~(size_t)255;
        return p;
    };
    float*    gi   = (float*)   alloc((size_t)SEQ * BATCH * G3 * sizeof(float));
    bf16_t*   xb   = (bf16_t*)  alloc((size_t)BATCH * SEQ * IN_DIM * sizeof(bf16_t));
    bf16_t*   wihb = (bf16_t*)  alloc((size_t)G3 * IN_DIM * sizeof(bf16_t));
    bf16_t*   whhb = (bf16_t*)  alloc((size_t)G3 * HID * sizeof(bf16_t));
    bf16_t*   hAb  = (bf16_t*)  alloc((size_t)BATCH * HID * sizeof(bf16_t));
    bf16_t*   hBb  = (bf16_t*)  alloc((size_t)BATCH * HID * sizeof(bf16_t));
    unsigned* bar  = (unsigned*)alloc(256);

    // bf16 conversions (layouts already match WMMA operand needs) + h0/bar init
    {
        size_t nx = (size_t)BATCH * SEQ * IN_DIM;
        f32_to_bf16_kernel<<<(unsigned)((nx + 255) / 256), 256, 0, stream>>>(x, xb, nx);
        size_t nw1 = (size_t)G3 * IN_DIM;
        f32_to_bf16_kernel<<<(unsigned)((nw1 + 255) / 256), 256, 0, stream>>>(W_ih, wihb, nw1);
        size_t nw2 = (size_t)G3 * HID;
        f32_to_bf16_kernel<<<(unsigned)((nw2 + 255) / 256), 256, 0, stream>>>(W_hh, whhb, nw2);
        int nh = BATCH * HID;
        init_kernel<<<(nh + 255) / 256, 256, 0, stream>>>(hAb, bar, nh);
    }

    // Phase 1: input projection for all timesteps
    {
        dim3 grid((BATCH * SEQ) / 32, G3 / 64);
        gi_gemm_kernel<<<grid, 256, 0, stream>>>(xb, wihb, b_ih, gi);
    }

    // Phase 2: persistent scan — ONE launch, 1024 grid-barrier-separated steps
    gru_scan_kernel<<<HID / 32, 256, 0, stream>>>(gi, whhb, b_hh,
                                                  hAb, hBb, out, bar);
}
